// EfficientSelfAttention_36747740184718
// MI455X (gfx1250) — compile-verified, matrix operand order checked by reading, not run on hardware
//
#include <hip/hip_runtime.h>
#include <hip/hip_bf16.h>
#include <math.h>

// ---------------------------------------------------------------------------
// Performer (FAVOR+) transformer block for MI455X (gfx1250, wave32, WMMA).
// Big GEMMs -> v_wmma_f32_16x16x32_bf16 (f32 accumulate), double-buffered
// fragment loads so VMEM latency overlaps the matrix pipe.
// Feature maps + causal linear-attention scan -> fp32 VALU (+ prefetch).
// ---------------------------------------------------------------------------

typedef __attribute__((ext_vector_type(16))) __bf16 v16bf;
typedef __attribute__((ext_vector_type(8)))  __bf16 v8bf;
typedef __attribute__((ext_vector_type(8)))  float  v8f;

#define B_   2
#define N_   2048
#define D_   768
#define H_   12
#define DH_  64
#define MF_  266
#define FF_  3072
#define R_   (B_ * N_)
#define BHN_ (B_ * H_ * N_)

// ---------------------------------------------------------------------------
// WMMA fragment load. Row-major [rows, K] bf16 source.
// A-layout (16x32 bf16): lanes 0-15 hold row r with K = kb+{0..7, 16..23},
// lanes 16-31 hold row r with K = kb+{8..15, 24..31}. B is loaded from a
// pre-transposed weight [Ncols, K] with the mirrored layout (N plays M).
// ---------------------------------------------------------------------------
__device__ __forceinline__ v16bf load_frag(const __bf16* base, int ld, int lane, int kb) {
  const int hl = lane >> 4;
  const int r  = lane & 15;
  const __bf16* p = base + (size_t)r * ld + kb + hl * 8;
  v8bf lo = *(const v8bf*)(p);
  v8bf hi = *(const v8bf*)(p + 16);
  v16bf out;
#pragma unroll
  for (int i = 0; i < 8; ++i) { out[i] = lo[i]; out[i + 8] = hi[i]; }
  return out;
}

// ---------------------------------------------------------------------------
// Generic WMMA GEMM: out[row,col] = A[row,:] . Bt[col,:] + bias[col]
// Block = 256 threads = 8 waves. Wave w -> rows [by*128 + 16w, +16),
// cols [bx*64, +64) via 4 accumulators. K-loop is register double-buffered:
// next step's A + 4 B fragments are issued before the current step's WMMAs,
// so s_wait_loadcnt covers loads that have a full WMMA group of latency
// hiding behind them. Epilogue: optional residual add, exact GELU, bf16 out.
// ---------------------------------------------------------------------------
__global__ __launch_bounds__(256) void wmma_gemm_kernel(
    const __bf16* __restrict__ A, const __bf16* __restrict__ Bt,
    const float* __restrict__ bias, const float* __restrict__ resid,
    float* __restrict__ outF, __bf16* __restrict__ outB,
    int K, int Ncols, int do_gelu) {
  const int lane = threadIdx.x & 31;
  const int wave = threadIdx.x >> 5;
  const int rowBase = blockIdx.y * 128 + wave * 16;
  const int colBase = blockIdx.x * 64;
  v8f acc[4] = {};
  const __bf16* Arow = A + (size_t)rowBase * K;
  const __bf16* B0 = Bt + (size_t)(colBase)      * K;
  const __bf16* B1 = Bt + (size_t)(colBase + 16) * K;
  const __bf16* B2 = Bt + (size_t)(colBase + 32) * K;
  const __bf16* B3 = Bt + (size_t)(colBase + 48) * K;

  // prologue: fragments for kb = 0
  v16bf a_cur  = load_frag(Arow, K, lane, 0);
  v16bf b_cur0 = load_frag(B0,   K, lane, 0);
  v16bf b_cur1 = load_frag(B1,   K, lane, 0);
  v16bf b_cur2 = load_frag(B2,   K, lane, 0);
  v16bf b_cur3 = load_frag(B3,   K, lane, 0);

  for (int kb = 32; kb < K; kb += 32) {
    // issue next step's loads first (latency hiding behind 4 WMMAs)
    v16bf a_nxt  = load_frag(Arow, K, lane, kb);
    v16bf b_nxt0 = load_frag(B0,   K, lane, kb);
    v16bf b_nxt1 = load_frag(B1,   K, lane, kb);
    v16bf b_nxt2 = load_frag(B2,   K, lane, kb);
    v16bf b_nxt3 = load_frag(B3,   K, lane, kb);
    acc[0] = __builtin_amdgcn_wmma_f32_16x16x32_bf16(false, a_cur, false, b_cur0, (short)0, acc[0], false, false);
    acc[1] = __builtin_amdgcn_wmma_f32_16x16x32_bf16(false, a_cur, false, b_cur1, (short)0, acc[1], false, false);
    acc[2] = __builtin_amdgcn_wmma_f32_16x16x32_bf16(false, a_cur, false, b_cur2, (short)0, acc[2], false, false);
    acc[3] = __builtin_amdgcn_wmma_f32_16x16x32_bf16(false, a_cur, false, b_cur3, (short)0, acc[3], false, false);
    a_cur = a_nxt; b_cur0 = b_nxt0; b_cur1 = b_nxt1; b_cur2 = b_nxt2; b_cur3 = b_nxt3;
  }
  // epilogue WMMA group
  acc[0] = __builtin_amdgcn_wmma_f32_16x16x32_bf16(false, a_cur, false, b_cur0, (short)0, acc[0], false, false);
  acc[1] = __builtin_amdgcn_wmma_f32_16x16x32_bf16(false, a_cur, false, b_cur1, (short)0, acc[1], false, false);
  acc[2] = __builtin_amdgcn_wmma_f32_16x16x32_bf16(false, a_cur, false, b_cur2, (short)0, acc[2], false, false);
  acc[3] = __builtin_amdgcn_wmma_f32_16x16x32_bf16(false, a_cur, false, b_cur3, (short)0, acc[3], false, false);

  // C/D layout: VGPR r, lanes 0-15 -> (M=r, N=lane), lanes 16-31 -> (M=r+8).
  const int hl = lane >> 4;
  const int c0 = lane & 15;
#pragma unroll
  for (int j = 0; j < 4; ++j) {
#pragma unroll
    for (int r = 0; r < 8; ++r) {
      int row = rowBase + hl * 8 + r;
      int col = colBase + j * 16 + c0;
      float v = acc[j][r] + bias[col];
      if (resid) v += resid[(size_t)row * Ncols + col];
      if (do_gelu) v = 0.5f * v * (1.0f + erff(v * 0.7071067811865475f));
      if (outB) outB[(size_t)row * Ncols + col] = (__bf16)v;
      else      outF[(size_t)row * Ncols + col] = v;
    }
  }
}

// Weight prep: float W[K,Ncols] -> bf16 Wt[Ncols,K] (transpose + convert).
__global__ void wprep_kernel(const float* __restrict__ W, __bf16* __restrict__ Wt,
                             int K, int Ncols) {
  size_t idx = (size_t)blockIdx.x * 256 + threadIdx.x;
  if (idx >= (size_t)K * Ncols) return;
  int n = (int)(idx / K);
  int k = (int)(idx % K);
  Wt[idx] = (__bf16)W[(size_t)k * Ncols + n];
}

// LayerNorm over D=768, one block per row, bf16 output for GEMM A.
__global__ __launch_bounds__(256) void ln_kernel(
    const float* __restrict__ x, const float* __restrict__ g,
    const float* __restrict__ bvec, __bf16* __restrict__ out) {
  __shared__ float red[256];
  const int t = threadIdx.x;
  const float* xr = x + (size_t)blockIdx.x * D_;
  float v0 = xr[t], v1 = xr[t + 256], v2 = xr[t + 512];
  red[t] = v0 + v1 + v2;
  __syncthreads();
  for (int s = 128; s > 0; s >>= 1) { if (t < s) red[t] += red[t + s]; __syncthreads(); }
  float mu = red[0] * (1.0f / D_);
  __syncthreads();
  float d0 = v0 - mu, d1 = v1 - mu, d2 = v2 - mu;
  red[t] = d0 * d0 + d1 * d1 + d2 * d2;
  __syncthreads();
  for (int s = 128; s > 0; s >>= 1) { if (t < s) red[t] += red[t + s]; __syncthreads(); }
  float rstd = rsqrtf(red[0] * (1.0f / D_) + 1e-5f);
  __bf16* orow = out + (size_t)blockIdx.x * D_;
  orow[t]       = (__bf16)(d0 * rstd * g[t]       + bvec[t]);
  orow[t + 256] = (__bf16)(d1 * rstd * g[t + 256] + bvec[t + 256]);
  orow[t + 512] = (__bf16)(d2 * rstd * g[t + 512] + bvec[t + 512]);
}

// Pass 1 for keys: per-row max of dd = dn*(k . proj_m).
__global__ __launch_bounds__(256) void featmax_kernel(
    const float* __restrict__ kf, const float* __restrict__ proj,
    float* __restrict__ rowmax) {
  __shared__ float x_s[DH_];
  __shared__ float red[256];
  const int t = threadIdx.x;
  const int n = blockIdx.x, h = blockIdx.y, b = blockIdx.z;
  const float* xr = kf + ((size_t)(b * N_ + n)) * D_ + h * DH_;
  if (t < DH_) x_s[t] = xr[t];
  __syncthreads();
  const float dn = 0.35355339059327373f;  // 64^-0.25
  float lm = -3.0e38f;
  for (int m = t; m < MF_; m += 256) {
    float dot = 0.f;
    const float* pr = proj + (size_t)m * DH_;
#pragma unroll
    for (int kk = 0; kk < DH_; ++kk) dot += x_s[kk] * pr[kk];
    lm = fmaxf(lm, dn * dot);
  }
  red[t] = lm;
  __syncthreads();
  for (int s = 128; s > 0; s >>= 1) { if (t < s) red[t] = fmaxf(red[t], red[t + s]); __syncthreads(); }
  if (t == 0) rowmax[(size_t)(b * H_ + h) * N_ + n] = red[0];
}

__global__ __launch_bounds__(256) void kmax_reduce_kernel(
    const float* __restrict__ rowmax, float* __restrict__ gmax) {
  __shared__ float red[256];
  const int t = threadIdx.x;
  float lm = -3.0e38f;
  for (int i = t; i < BHN_; i += 256) lm = fmaxf(lm, rowmax[i]);
  red[t] = lm;
  __syncthreads();
  for (int s = 128; s > 0; s >>= 1) { if (t < s) red[t] = fmaxf(red[t], red[t + s]); __syncthreads(); }
  if (t == 0) gmax[0] = red[0];
}

// FAVOR+ positive softmax features: ratio*(exp(dd - diag - mx) + 1e-4).
__global__ __launch_bounds__(256) void feat_kernel(
    const float* __restrict__ data, const float* __restrict__ proj,
    float* __restrict__ out, const float* __restrict__ gmaxp, int is_query) {
  __shared__ float x_s[DH_];
  __shared__ float dd_s[MF_];
  __shared__ float red[256];
  const int t = threadIdx.x;
  const int n = blockIdx.x, h = blockIdx.y, b = blockIdx.z;
  const float* xr = data + ((size_t)(b * N_ + n)) * D_ + h * DH_;
  if (t < DH_) x_s[t] = xr[t];
  __syncthreads();
  const float dn    = 0.35355339059327373f;   // 64^-0.25
  const float ratio = 0.06131393394849658f;   // 266^-0.5
  float ps = (t < DH_) ? x_s[t] * x_s[t] : 0.f;
  red[t] = ps;
  __syncthreads();
  for (int s = 128; s > 0; s >>= 1) { if (t < s) red[t] += red[t + s]; __syncthreads(); }
  float diag = 0.5f * dn * dn * red[0];
  __syncthreads();
  float lm = -3.0e38f;
  for (int m = t; m < MF_; m += 256) {
    float dot = 0.f;
    const float* pr = proj + (size_t)m * DH_;
#pragma unroll
    for (int kk = 0; kk < DH_; ++kk) dot += x_s[kk] * pr[kk];
    float dd = dn * dot;
    dd_s[m] = dd;
    lm = fmaxf(lm, dd);
  }
  float mx;
  if (is_query) {
    red[t] = lm;
    __syncthreads();
    for (int s = 128; s > 0; s >>= 1) { if (t < s) red[t] = fmaxf(red[t], red[t + s]); __syncthreads(); }
    mx = red[0];
  } else {
    mx = gmaxp[0];
  }
  float* orow = out + ((size_t)(b * H_ + h) * N_ + n) * MF_;
  for (int m = t; m < MF_; m += 256)
    orow[m] = ratio * (expf(dd_s[m] - diag - mx) + 1e-4f);
}

// ---------------------------------------------------------------------------
// Causal linear attention scan. One block per (b,h). State S[266][64] and
// z[266] live in registers, striped: thread t owns e = t&63, m = (t>>6) mod 4.
// Per row n (inclusive update, matching the reference cumsum):
//   S[m][e] += kp[n,m]*v[n,e];  out[e] = sum_m qp[n,m]*S[m][e] / den
//   den = sum_m qp[n,m]*(z[m] + 1e-6),  z[m] += kp[n,m]
// Next row's operands are prefetched (global_prefetch_b8) during compute.
// ---------------------------------------------------------------------------
__global__ __launch_bounds__(256) void attn_kernel(
    const float* __restrict__ qp, const float* __restrict__ kp,
    const float* __restrict__ vf, __bf16* __restrict__ o) {
  __shared__ float kp_s[MF_], qp_s[MF_], v_s[DH_];
  __shared__ float ored[4][DH_];
  __shared__ float dred[4];
  const int t  = threadIdx.x;
  const int e  = t & 63;
  const int mg = t >> 6;  // 0..3
  const int h = blockIdx.x, b = blockIdx.y;
  const float* qpb = qp + (size_t)(b * H_ + h) * N_ * MF_;
  const float* kpb = kp + (size_t)(b * H_ + h) * N_ * MF_;
  const float* vb  = vf + (size_t)(b * N_) * D_ + h * DH_;
  __bf16* ob = o + (size_t)(b * N_) * D_ + h * DH_;
  const int MJ = 67;  // ceil(266/4)
  float S_loc[MJ], z_loc[MJ];
#pragma unroll
  for (int j = 0; j < MJ; ++j) { S_loc[j] = 0.f; z_loc[j] = 0.f; }
  for (int n = 0; n < N_; ++n) {
    for (int i = t; i < MF_; i += 256) {
      kp_s[i] = kpb[(size_t)n * MF_ + i];
      qp_s[i] = qpb[(size_t)n * MF_ + i];
    }
    if (t < DH_) v_s[t] = vb[(size_t)n * D_ + t];
    // prefetch next row while this row computes (lowers to global_prefetch_b8)
    if (n + 1 < N_) {
      if (t < 2) {
        __builtin_prefetch(kpb + (size_t)(n + 1) * MF_ + t * 128, 0, 3);
        __builtin_prefetch(qpb + (size_t)(n + 1) * MF_ + t * 128, 0, 3);
      }
      if (t == 2) __builtin_prefetch(vb + (size_t)(n + 1) * D_, 0, 3);
    }
    __syncthreads();
    float ve = v_s[e];
    float po = 0.f, pd = 0.f;
#pragma unroll
    for (int j = 0; j < MJ; ++j) {
      int m = mg + 4 * j;
      if (m < MF_) {
        float kk = kp_s[m];
        float qq = qp_s[m];
        S_loc[j] += kk * ve;
        po += qq * S_loc[j];
        if (e == 0) { z_loc[j] += kk; pd += qq * (z_loc[j] + 1e-6f); }
      }
    }
    ored[mg][e] = po;
    if (e == 0) dred[mg] = pd;
    __syncthreads();
    if (mg == 0) {
      float den = dred[0] + dred[1] + dred[2] + dred[3];
      float oo = (ored[0][e] + ored[1][e] + ored[2][e] + ored[3][e]) / den;
      ob[(size_t)n * D_ + e] = (__bf16)oo;
    }
    __syncthreads();
  }
}

// ---------------------------------------------------------------------------
extern "C" void kernel_launch(void* const* d_in, const int* in_sizes, int n_in,
                              void* d_out, int out_size, void* d_ws, size_t ws_size,
                              hipStream_t stream) {
  const float* x     = (const float*)d_in[0];
  const float* ln1_g = (const float*)d_in[1];
  const float* ln1_b = (const float*)d_in[2];
  const float* Wq    = (const float*)d_in[3];
  const float* bq    = (const float*)d_in[4];
  const float* Wk    = (const float*)d_in[5];
  const float* bk    = (const float*)d_in[6];
  const float* Wv    = (const float*)d_in[7];
  const float* bv    = (const float*)d_in[8];
  const float* Wo    = (const float*)d_in[9];
  const float* bo    = (const float*)d_in[10];
  const float* proj  = (const float*)d_in[11];
  const float* ln2_g = (const float*)d_in[12];
  const float* ln2_b = (const float*)d_in[13];
  const float* W1    = (const float*)d_in[14];
  const float* b1    = (const float*)d_in[15];
  const float* W2    = (const float*)d_in[16];
  const float* b2    = (const float*)d_in[17];
  float* yout = (float*)d_out;
  (void)in_sizes; (void)n_in; (void)out_size; (void)ws_size;

  char* p = (char*)d_ws;
  auto carve = [&](size_t bytes) -> void* {
    void* r = (void*)p;
    p += (bytes + 255) & ~(size_t)255;
    return r;
  };
  __bf16* h_bf   = (__bf16*)carve((size_t)R_ * D_ * 2);
  __bf16* Wq_t   = (__bf16*)carve((size_t)D_ * D_ * 2);
  __bf16* Wk_t   = (__bf16*)carve((size_t)D_ * D_ * 2);
  __bf16* Wv_t   = (__bf16*)carve((size_t)D_ * D_ * 2);
  __bf16* Wo_t   = (__bf16*)carve((size_t)D_ * D_ * 2);
  __bf16* W1_t   = (__bf16*)carve((size_t)D_ * FF_ * 2);   // [FF, D]
  __bf16* W2_t   = (__bf16*)carve((size_t)FF_ * D_ * 2);   // [D, FF]
  float*  qf     = (float*)carve((size_t)R_ * D_ * 4);
  float*  kf     = (float*)carve((size_t)R_ * D_ * 4);
  float*  vfb    = (float*)carve((size_t)R_ * D_ * 4);
  float*  qp     = (float*)carve((size_t)BHN_ * MF_ * 4);
  float*  kp     = (float*)carve((size_t)BHN_ * MF_ * 4);
  float*  rowmax = (float*)carve((size_t)BHN_ * 4);
  float*  gmax   = (float*)carve(256);
  __bf16* o_bf   = (__bf16*)carve((size_t)R_ * D_ * 2);
  float*  x1     = (float*)carve((size_t)R_ * D_ * 4);
  __bf16* h2_bf  = (__bf16*)carve((size_t)R_ * D_ * 2);
  __bf16* a1_bf  = (__bf16*)carve((size_t)R_ * FF_ * 2);

  // 1) weight prep (float [K,N] -> bf16 [N,K])
  {
    int gdd = (D_ * D_ + 255) / 256;
    int gdf = (D_ * FF_ + 255) / 256;
    wprep_kernel<<<gdd, 256, 0, stream>>>(Wq, Wq_t, D_, D_);
    wprep_kernel<<<gdd, 256, 0, stream>>>(Wk, Wk_t, D_, D_);
    wprep_kernel<<<gdd, 256, 0, stream>>>(Wv, Wv_t, D_, D_);
    wprep_kernel<<<gdd, 256, 0, stream>>>(Wo, Wo_t, D_, D_);
    wprep_kernel<<<gdf, 256, 0, stream>>>(W1, W1_t, D_, FF_);   // Wt [FF, D]
    wprep_kernel<<<gdf, 256, 0, stream>>>(W2, W2_t, FF_, D_);   // Wt [D, FF]
  }

  // 2) LN1
  ln_kernel<<<R_, 256, 0, stream>>>(x, ln1_g, ln1_b, h_bf);

  // 3) QKV GEMMs (WMMA)
  dim3 gemmD(D_ / 64, R_ / 128);
  wmma_gemm_kernel<<<gemmD, 256, 0, stream>>>(h_bf, Wq_t, bq, nullptr, qf, nullptr, D_, D_, 0);
  wmma_gemm_kernel<<<gemmD, 256, 0, stream>>>(h_bf, Wk_t, bk, nullptr, kf, nullptr, D_, D_, 0);
  wmma_gemm_kernel<<<gemmD, 256, 0, stream>>>(h_bf, Wv_t, bv, nullptr, vfb, nullptr, D_, D_, 0);

  // 4) global max over key features
  dim3 featG(N_, H_, B_);
  featmax_kernel<<<featG, 256, 0, stream>>>(kf, proj, rowmax);
  kmax_reduce_kernel<<<1, 256, 0, stream>>>(rowmax, gmax);

  // 5) FAVOR+ feature maps
  feat_kernel<<<featG, 256, 0, stream>>>(qf, proj, qp, gmax, 1);
  feat_kernel<<<featG, 256, 0, stream>>>(kf, proj, kp, gmax, 0);

  // 6) causal linear attention scan
  attn_kernel<<<dim3(H_, B_), 256, 0, stream>>>(qp, kp, vfb, o_bf);

  // 7) x1 = x + o @ Wo + bo
  wmma_gemm_kernel<<<gemmD, 256, 0, stream>>>(o_bf, Wo_t, bo, x, x1, nullptr, D_, D_, 0);

  // 8) LN2
  ln_kernel<<<R_, 256, 0, stream>>>(x1, ln2_g, ln2_b, h2_bf);

  // 9) FFN1: a1 = gelu(h2 @ W1 + b1) -> bf16
  dim3 gemmF1(FF_ / 64, R_ / 128);
  wmma_gemm_kernel<<<gemmF1, 256, 0, stream>>>(h2_bf, W1_t, b1, nullptr, nullptr, a1_bf, D_, FF_, 1);

  // 10) FFN2: y = x1 + a1 @ W2 + b2 -> d_out
  wmma_gemm_kernel<<<gemmD, 256, 0, stream>>>(a1_bf, W2_t, b2, x1, yout, nullptr, FF_, D_, 0);
}
